// GIFT4Rec_advanced_mf_70643622084958
// MI455X (gfx1250) — matrix operand order repaired and checked by reading, not compiled, not run on hardware
//
#include <hip/hip_runtime.h>
#include <cstddef>

typedef __attribute__((ext_vector_type(16))) _Float16 v16h;
typedef __attribute__((ext_vector_type(8)))  _Float16 v8h;
typedef __attribute__((ext_vector_type(8)))  float    v8f;

#define DEMB 128
#define HDIM 256   // H = 2*D
#define NCATF 8
#define NNUMF 16
#define NTW 4      // N-tiles per wave in the WMMA GEMM

__device__ __forceinline__ float sigmf(float x) { return 1.0f / (1.0f + expf(-x)); }

// Hardware float atomic-add at device scope (L2 atomic units). Non-returning
// form -> tracked with STOREcnt; S_ENDPGM's implicit wait-idle covers kernel
// exit, and kernel-boundary ordering covers consumers.
__device__ __forceinline__ void atomAddF(float* p, float v) {
  asm volatile("global_atomic_add_f32 %0, %1, off scope:SCOPE_DEV"
               :: "v"((unsigned long long)p), "v"(v)
               : "memory");
}

// ---------------------------------------------------------------------------
// Precompute masked weights: Wm = (f16)(W * sigmoid(Mw[task])), biases in f32.
// ---------------------------------------------------------------------------
__global__ void k_prep_weights(const float* __restrict__ W0, const float* __restrict__ Mw0,
                               const float* __restrict__ b0, const float* __restrict__ Mb0,
                               const float* __restrict__ W1, const float* __restrict__ Mw1,
                               const float* __restrict__ b1, const float* __restrict__ Mb1,
                               const float* __restrict__ W2, const float* __restrict__ Mw2,
                               const float* __restrict__ b2, const float* __restrict__ Mb2,
                               _Float16* __restrict__ W0h, float* __restrict__ b0m,
                               _Float16* __restrict__ W1h, float* __restrict__ b1m,
                               float* __restrict__ w2m, float* __restrict__ b2m) {
  int tid = blockIdx.x * blockDim.x + threadIdx.x;
  const int perTask = HDIM * HDIM;             // 65536
  if (tid < 2 * perTask) {
    int jk = tid % perTask;                    // Mw is [2][H][2D], W is [H][2D]
    W0h[tid] = (_Float16)(W0[jk] * sigmf(Mw0[tid]));
    W1h[tid] = (_Float16)(W1[jk] * sigmf(Mw1[tid]));
  }
  if (tid < 2 * HDIM) {
    int j = tid % HDIM;
    b0m[tid] = b0[j] * sigmf(Mb0[tid]);
    b1m[tid] = b1[j] * sigmf(Mb1[tid]);
    w2m[tid] = W2[j] * sigmf(Mw2[tid]);        // W2 is [1][H]
  }
  if (tid < 2) b2m[tid] = b2[0] * sigmf(Mb2[tid]);
}

// ---------------------------------------------------------------------------
__global__ void k_clear(int* __restrict__ mark, float* __restrict__ agg,
                        int n_nodes, int agg_elems) {
  int tid = blockIdx.x * blockDim.x + threadIdx.x;
  if (tid < n_nodes) mark[tid] = -1;
  if (tid < agg_elems) agg[tid] = 0.0f;
}

// ---------------------------------------------------------------------------
// Claim a compact agg slot per needed global node id (CAS winner's tid = slot).
// ---------------------------------------------------------------------------
__global__ void k_claim(const int* __restrict__ user_ids, const int* __restrict__ item_ids,
                        int* __restrict__ mark, int B, int n_ent) {
  int tid = blockIdx.x * blockDim.x + threadIdx.x;
  if (tid >= 2 * B) return;
  int g = (tid < B) ? (user_ids[tid] + n_ent) : item_ids[tid - B];
  atomicCAS(&mark[g], -1, tid);
}

// ---------------------------------------------------------------------------
// Ballot-compacted edge scatter. Each lane filters one edge (coalesced loads
// of A_rows + mark); the wave then cooperatively processes only the hit edges:
// float4 gather of the 512B source row + hardware float atomics into agg.
// ---------------------------------------------------------------------------
__global__ void k_edge_scatter(const int* __restrict__ A_rows, const int* __restrict__ A_cols,
                               const float* __restrict__ A_vals,
                               const float* __restrict__ ent, const int* __restrict__ mark,
                               float* __restrict__ agg, int E) {
  int lane = threadIdx.x & 31;
  int wid  = (blockIdx.x * blockDim.x + threadIdx.x) >> 5;
  int nw   = (gridDim.x * blockDim.x) >> 5;
  for (int base = wid * 32; base < E; base += nw * 32) {
    int e = base + lane;
    int s = -1;
    if (e < E) s = mark[A_rows[e]];
    unsigned long long m = __ballot(s >= 0);       // wave32: low 32 bits
    while (m) {
      int src = __ffsll(m) - 1;
      m &= m - 1;
      int sh = __shfl(s, src, 32);                 // claimed slot (broadcast)
      int eh = base + src;                         // hit edge index
      float v = A_vals[eh];
      int   c = A_cols[eh];
      const float4 x = *(const float4*)(ent + (size_t)c * DEMB + lane * 4);
      float* dst = agg + (size_t)sh * DEMB + lane * 4;
      atomAddF(dst + 0, v * x.x);
      atomAddF(dst + 1, v * x.y);
      atomAddF(dst + 2, v * x.z);
      atomAddF(dst + 3, v * x.w);
    }
  }
}

// ---------------------------------------------------------------------------
// embed = ent[g] + agg[slot]; feat = cat-mean + relu(num @ numW.T + numb).
// Writes f32 embed/feat (blend) and f16 x = [embed | feat] (WMMA input).
// One 128-thread block per row; thread = dim d.
// ---------------------------------------------------------------------------
__global__ void k_build_x(const float* __restrict__ ent, const float* __restrict__ agg,
                          const int* __restrict__ mark,
                          const int* __restrict__ ids, int id_off,
                          const int* __restrict__ cat_feats, const float* __restrict__ cat_table,
                          const float* __restrict__ num_feats, const float* __restrict__ numW,
                          const float* __restrict__ numb,
                          float* __restrict__ embed_out, float* __restrict__ feat_out,
                          _Float16* __restrict__ x_out) {
  int b = blockIdx.x;
  int d = threadIdx.x;                       // 0..127
  int local = ids[b];
  int g = local + id_off;

  float e = ent[(size_t)g * DEMB + d];
  int s = mark[g];
  if (s >= 0) e += agg[(size_t)s * DEMB + d];

  float csum = 0.0f; int cnt = 0;
#pragma unroll
  for (int f = 0; f < NCATF; ++f) {
    int cid = cat_feats[(size_t)local * NCATF + f];
    cnt += (cid > 0) ? 1 : 0;
    csum += cat_table[(size_t)cid * DEMB + d];   // row 0 is all zeros
  }
  float feat = (cnt > 0) ? csum * (1.0f / (float)cnt) : 0.0f;

  float acc = numb[d];
#pragma unroll
  for (int k = 0; k < NNUMF; ++k)
    acc += num_feats[(size_t)local * NNUMF + k] * numW[(size_t)d * NNUMF + k];
  feat += fmaxf(acc, 0.0f);

  embed_out[(size_t)b * DEMB + d] = e;
  feat_out[(size_t)b * DEMB + d]  = feat;
  x_out[(size_t)b * HDIM + d]        = (_Float16)e;
  x_out[(size_t)b * HDIM + DEMB + d] = (_Float16)feat;
}

// ---------------------------------------------------------------------------
// Y[B,256] = relu(X[B,256] @ W^T + bias), f16 operands, f32 accumulate.
// One wave -> 4 adjacent 16x16 N-tiles (shared A fragment, 4 independent
// accumulators -> back-to-back independent v_wmma issues).
// Fragment packing used here (contiguous k, same order for A and B):
//   A: lanes 0-15 row M=lane, halves K 0..15; lanes 16-31 halves K 16..31
//   B: lanes 0-15 col N=lane, halves K 0..15; lanes 16-31 halves K 16..31
//   D: acc[r] -> row r + 8*(lane>>4), col lane&15
// ---------------------------------------------------------------------------
__global__ void k_gemm_relu(const _Float16* __restrict__ X, const _Float16* __restrict__ W,
                            const float* __restrict__ bias, _Float16* __restrict__ Y,
                            int Bsz) {
  const int K = HDIM, NF = HDIM;
  const int ngroups = NF / (16 * NTW);                       // 4
  int wave = (blockIdx.x * blockDim.x + threadIdx.x) >> 5;
  int lane = threadIdx.x & 31;
  int mt = wave / ngroups;
  int ng = wave - mt * ngroups;
  if (mt * 16 >= Bsz) return;                                // wave-uniform

  int laneLo = lane & 15;
  int laneHi = lane >> 4;
  const _Float16* xrow  = X + (size_t)(mt * 16 + laneLo) * K;
  const _Float16* wbase = W + (size_t)(ng * (16 * NTW) + laneLo) * K;
  const int kbaseA = laneHi * 16;
  const int kbaseB = laneHi * 16;

  v8f acc[NTW] = {};
#pragma unroll
  for (int k0 = 0; k0 < K; k0 += 32) {
    v8h alo = *(const v8h*)(xrow + k0 + kbaseA);
    v8h ahi = *(const v8h*)(xrow + k0 + kbaseA + 8);
    v16h a;
#pragma unroll
    for (int i = 0; i < 8; ++i) { a[i] = alo[i]; a[i + 8] = ahi[i]; }
#pragma unroll
    for (int j = 0; j < NTW; ++j) {
      v16h b = *(const v16h*)(wbase + (size_t)j * 16 * K + k0 + kbaseB);
      acc[j] = __builtin_amdgcn_wmma_f32_16x16x32_f16(
          /*neg_a=*/false, a, /*neg_b=*/false, b,
          /*c_mod=*/(short)0, acc[j], /*reuse_a=*/false, /*reuse_b=*/false);
    }
  }

#pragma unroll
  for (int j = 0; j < NTW; ++j) {
    int nn = ng * (16 * NTW) + j * 16 + laneLo;
    float bb = bias[nn];
#pragma unroll
    for (int r = 0; r < 8; ++r) {
      int mm = mt * 16 + r + 8 * laneHi;
      float v = acc[j][r] + bb;
      Y[(size_t)mm * NF + nn] = (_Float16)(v > 0.0f ? v : 0.0f);
    }
  }
}

// ---------------------------------------------------------------------------
// w[b] = sigmoid(h1[b,:] . w2m + b2m[0]) — one wave per row.
// ---------------------------------------------------------------------------
__global__ void k_wscalar(const _Float16* __restrict__ h1, const float* __restrict__ w2m,
                          const float* __restrict__ b2m_task, float* __restrict__ wout, int B) {
  int wid = (blockIdx.x * blockDim.x + threadIdx.x) >> 5;
  int lane = threadIdx.x & 31;
  if (wid >= B) return;
  const _Float16* row = h1 + (size_t)wid * HDIM;
  float s = 0.0f;
#pragma unroll
  for (int k = lane; k < HDIM; k += 32) s += (float)row[k] * w2m[k];
#pragma unroll
  for (int off = 16; off > 0; off >>= 1) s += __shfl_xor(s, off, 32);
  if (lane == 0) wout[wid] = sigmf(s + b2m_task[0]);
}

// ---------------------------------------------------------------------------
// out[b] = sum_d (wu*ue + (1-wu)*uf) * (wi*ie + (1-wi)*if) — one wave per row.
// ---------------------------------------------------------------------------
__global__ void k_final(const float* __restrict__ ue, const float* __restrict__ uf,
                        const float* __restrict__ ie, const float* __restrict__ iff,
                        const float* __restrict__ wu, const float* __restrict__ wi,
                        float* __restrict__ out, int B) {
  int wid = (blockIdx.x * blockDim.x + threadIdx.x) >> 5;
  int lane = threadIdx.x & 31;
  if (wid >= B) return;
  float a = wu[wid], c = wi[wid];
  float s = 0.0f;
#pragma unroll
  for (int d = lane; d < DEMB; d += 32) {
    size_t idx = (size_t)wid * DEMB + d;
    float u = a * ue[idx] + (1.0f - a) * uf[idx];
    float v = c * ie[idx] + (1.0f - c) * iff[idx];
    s += u * v;
  }
#pragma unroll
  for (int off = 16; off > 0; off >>= 1) s += __shfl_xor(s, off, 32);
  if (lane == 0) out[wid] = s;
}

// ---------------------------------------------------------------------------
extern "C" void kernel_launch(void* const* d_in, const int* in_sizes, int n_in,
                              void* d_out, int out_size, void* d_ws, size_t ws_size,
                              hipStream_t stream) {
  const float* ent      = (const float*)d_in[0];
  const float* A_vals   = (const float*)d_in[1];
  const float* u_table  = (const float*)d_in[2];
  const float* i_table  = (const float*)d_in[3];
  const float* numW_u   = (const float*)d_in[4];
  const float* numb_u   = (const float*)d_in[5];
  const float* numW_i   = (const float*)d_in[6];
  const float* numb_i   = (const float*)d_in[7];
  const float* W0  = (const float*)d_in[8];
  const float* b0  = (const float*)d_in[9];
  const float* Mw0 = (const float*)d_in[10];
  const float* Mb0 = (const float*)d_in[11];
  const float* W1  = (const float*)d_in[12];
  const float* b1  = (const float*)d_in[13];
  const float* Mw1 = (const float*)d_in[14];
  const float* Mb1 = (const float*)d_in[15];
  const float* W2  = (const float*)d_in[16];
  const float* b2  = (const float*)d_in[17];
  const float* Mw2 = (const float*)d_in[18];
  const float* Mb2 = (const float*)d_in[19];
  const float* u_num  = (const float*)d_in[20];
  const float* i_num  = (const float*)d_in[21];
  const int* A_rows   = (const int*)d_in[22];
  const int* A_cols   = (const int*)d_in[23];
  const int* u_cat    = (const int*)d_in[24];
  const int* i_cat    = (const int*)d_in[25];
  const int* user_ids = (const int*)d_in[26];
  const int* item_ids = (const int*)d_in[27];

  const int N     = in_sizes[0] / DEMB;      // 250000
  const int E     = in_sizes[1];             // 2000000
  const int B     = in_sizes[26];            // 8192
  const int N_ENT = in_sizes[21] / NNUMF;    // 200000
  (void)ws_size; (void)n_in; (void)out_size;

  // ---- bump allocator on d_ws (~52 MB total) ----
  char* wsb = (char*)d_ws;
  size_t off = 0;
  auto alloc = [&](size_t bytes) -> void* {
    off = (off + 255) & ~(size_t)255;
    void* p = wsb + off;
    off += bytes;
    return p;
  };

  int*      mark = (int*)alloc((size_t)N * sizeof(int));
  float*    agg  = (float*)alloc((size_t)2 * B * DEMB * sizeof(float));
  _Float16* W0h  = (_Float16*)alloc((size_t)2 * HDIM * HDIM * sizeof(_Float16));
  _Float16* W1h  = (_Float16*)alloc((size_t)2 * HDIM * HDIM * sizeof(_Float16));
  float*    b0m  = (float*)alloc(2 * HDIM * sizeof(float));
  float*    b1m  = (float*)alloc(2 * HDIM * sizeof(float));
  float*    w2m  = (float*)alloc(2 * HDIM * sizeof(float));
  float*    b2m  = (float*)alloc(2 * sizeof(float));

  float*    u_embed = (float*)alloc((size_t)B * DEMB * sizeof(float));
  float*    u_feat  = (float*)alloc((size_t)B * DEMB * sizeof(float));
  _Float16* u_x0    = (_Float16*)alloc((size_t)B * HDIM * sizeof(_Float16));
  _Float16* u_h0    = (_Float16*)alloc((size_t)B * HDIM * sizeof(_Float16));
  _Float16* u_h1    = (_Float16*)alloc((size_t)B * HDIM * sizeof(_Float16));
  float*    i_embed = (float*)alloc((size_t)B * DEMB * sizeof(float));
  float*    i_feat  = (float*)alloc((size_t)B * DEMB * sizeof(float));
  _Float16* i_x0    = (_Float16*)alloc((size_t)B * HDIM * sizeof(_Float16));
  _Float16* i_h0    = (_Float16*)alloc((size_t)B * HDIM * sizeof(_Float16));
  _Float16* i_h1    = (_Float16*)alloc((size_t)B * HDIM * sizeof(_Float16));
  float*    wu      = (float*)alloc((size_t)B * sizeof(float));
  float*    wi      = (float*)alloc((size_t)B * sizeof(float));

  {
    int total = 2 * HDIM * HDIM;
    k_prep_weights<<<(total + 255) / 256, 256, 0, stream>>>(
        W0, Mw0, b0, Mb0, W1, Mw1, b1, Mb1, W2, Mw2, b2, Mb2,
        W0h, b0m, W1h, b1m, w2m, b2m);
  }
  {
    int elems = 2 * B * DEMB;                 // > N
    k_clear<<<(elems + 255) / 256, 256, 0, stream>>>(mark, agg, N, elems);
  }
  k_claim<<<(2 * B + 255) / 256, 256, 0, stream>>>(user_ids, item_ids, mark, B, N_ENT);
  k_edge_scatter<<<2048, 256, 0, stream>>>(A_rows, A_cols, A_vals, ent, mark, agg, E);
  k_build_x<<<B, DEMB, 0, stream>>>(ent, agg, mark, user_ids, N_ENT,
                                    u_cat, u_table, u_num, numW_u, numb_u,
                                    u_embed, u_feat, u_x0);
  k_build_x<<<B, DEMB, 0, stream>>>(ent, agg, mark, item_ids, 0,
                                    i_cat, i_table, i_num, numW_i, numb_i,
                                    i_embed, i_feat, i_x0);
  {
    int waves  = (B / 16) * (HDIM / (16 * NTW));  // 512 * 4 = 2048
    int blocks = (waves + 7) / 8;                 // 8 waves per 256-thread block
    const size_t wOff = (size_t)HDIM * HDIM;      // per-task weight stride
    k_gemm_relu<<<blocks, 256, 0, stream>>>(u_x0, W0h,        b0m,        u_h0, B);
    k_gemm_relu<<<blocks, 256, 0, stream>>>(u_h0, W1h,        b1m,        u_h1, B);
    k_gemm_relu<<<blocks, 256, 0, stream>>>(i_x0, W0h + wOff, b0m + HDIM, i_h0, B);
    k_gemm_relu<<<blocks, 256, 0, stream>>>(i_h0, W1h + wOff, b1m + HDIM, i_h1, B);
  }
  k_wscalar<<<(B * 32 + 255) / 256, 256, 0, stream>>>(u_h1, w2m,        b2m,     wu, B);
  k_wscalar<<<(B * 32 + 255) / 256, 256, 0, stream>>>(i_h1, w2m + HDIM, b2m + 1, wi, B);
  k_final<<<(B * 32 + 255) / 256, 256, 0, stream>>>(u_embed, u_feat, i_embed, i_feat,
                                                    wu, wi, (float*)d_out, B);
}
